// MHIM_87917980549599
// MI455X (gfx1250) — compile-verified, matrix operand order checked by reading, not run on hardware
//
#include <hip/hip_runtime.h>
#include <math.h>

// ---------------------------------------------------------------------------
// MI455X (gfx1250) implementation.
//
// Roofline: reference is 110 GFLOP fp32 (kv GEMM dominated). Since KQ=10 <<
// C=1024, we refactor:  dots = (q Wk^T) @ xn^T  and  out = ((w @ xn) @ Wv),
// reducing heavy compute to ~9 GFLOP, executed with v_wmma_f32_16x16x32_bf16
// (wave32 WMMA). Intermediates (xn^T bf16 54MB, dots 8.4MB, w 4.2MB) fit in
// the 192MB L2, so the GEMM passes stream from L2, and HBM traffic stays near
// the ~170MB floor (x once, weights, 27MB output).
// ---------------------------------------------------------------------------

#define L_TOK   32768
#define C_DIM   1024
#define N_H     8
#define D_H     128
#define K_Q     10
#define KSEL    26215          // ceil(L * 0.8)
#define LKEEP   6553           // L - KSEL
#define NPAD    26240          // 410 * 64, multiple of 64 (and of 32 for WMMA K)
#define LN_EPS  1e-5f
#define QK_SCALE 0.08838834764831845f   // 128^-0.5
#define STRH    1032           // LDS row stride in bf16 halves (2064 B: 16B-aligned, 4-bank rotation)

typedef __attribute__((ext_vector_type(4)))  unsigned int u32x4;
typedef __attribute__((ext_vector_type(8)))  float        v8f;
typedef __attribute__((ext_vector_type(16))) __bf16       v16bf;

union FragU { struct { u32x4 a, b; } u; v16bf f; };

// Monotonic key for float ordering (handles sign bit).
__device__ __forceinline__ unsigned mapkey(float f) {
  unsigned u = __float_as_uint(f);
  return u ^ ((u & 0x80000000u) ? 0xFFFFFFFFu : 0x80000000u);
}

// ---------------------------------------------------------------------------
// Kernel 1: per-head exact radix select of the LKEEP-th largest attn value.
// Elements strictly below it are the KSEL smallest -> "voted".
// Also zeroes the vote histogram (block 0).
// ---------------------------------------------------------------------------
__global__ void k_radix(const float* __restrict__ attn, unsigned* __restrict__ thresh,
                        int* __restrict__ c_hist) {
  const int h = blockIdx.x;
  __shared__ unsigned hist[256];
  __shared__ unsigned s_prefix;
  __shared__ int s_k;
  if (threadIdx.x == 0) { s_prefix = 0u; s_k = LKEEP; }
  if (h == 0 && threadIdx.x < 16) c_hist[threadIdx.x] = 0;
  const float* ar = attn + (size_t)h * L_TOK;
  for (int byte = 3; byte >= 0; --byte) {
    hist[threadIdx.x] = 0u;
    __syncthreads();
    const unsigned pfx = s_prefix;
    const int sh_hi = (byte + 1) * 8;
    for (int i = threadIdx.x; i < L_TOK; i += 256) {
      unsigned key = mapkey(ar[i]);
      bool match = (byte == 3) || ((key >> sh_hi) == (pfx >> sh_hi));
      if (match) atomicAdd(&hist[(key >> (byte * 8)) & 0xFF], 1u);
    }
    __syncthreads();
    if (threadIdx.x == 0) {
      int cum = 0, k = s_k;
      for (int bb = 255; bb >= 0; --bb) {
        int c = (int)hist[bb];
        if (cum + c >= k) { s_prefix = pfx | ((unsigned)bb << (byte * 8)); s_k = k - cum; break; }
        cum += c;
      }
    }
    __syncthreads();
  }
  if (threadIdx.x == 0) thresh[h] = s_prefix;
}

// ---------------------------------------------------------------------------
// Kernel 2: vote[l] = #heads voting for l; histogram of vote values.
// ---------------------------------------------------------------------------
__global__ void k_vote(const float* __restrict__ attn, const unsigned* __restrict__ thresh,
                       int* __restrict__ vote, int* __restrict__ c_hist) {
  const int i = blockIdx.x * 256 + threadIdx.x;
  int cnt = 0;
  #pragma unroll
  for (int h = 0; h < N_H; ++h)
    cnt += (mapkey(attn[h * L_TOK + i]) < thresh[h]) ? 1 : 0;
  vote[i] = cnt;
  atomicAdd(&c_hist[cnt], 1);
}

// ---------------------------------------------------------------------------
// Kernel 3: single-block stable compaction reproducing jax top_k tie-break +
// stable argsort of the 0/1 mask. Produces ids_keep / ids_masked.
// ---------------------------------------------------------------------------
__device__ int block_scan_inc(int v, int* sd) {
  int t = threadIdx.x;
  sd[t] = v; __syncthreads();
  for (int off = 1; off < 1024; off <<= 1) {
    int u = (t >= off) ? sd[t - off] : 0;
    __syncthreads();
    sd[t] += u;
    __syncthreads();
  }
  int r = sd[t];
  __syncthreads();
  return r;
}

__global__ void k_compact(const int* __restrict__ vote, const int* __restrict__ c_hist,
                          int* __restrict__ ids_keep, int* __restrict__ ids_masked) {
  __shared__ int sd[1024];
  __shared__ int s_vt, s_m, s_ceq, s_cm;
  if (threadIdx.x == 0) {
    int cum = 0, vt = 0, m = 0;
    for (int v = 15; v >= 0; --v) {
      int c = c_hist[v];
      if (cum + c >= KSEL) { vt = v; m = KSEL - cum; break; }
      cum += c;
    }
    s_vt = vt; s_m = m; s_ceq = 0; s_cm = 0;
  }
  __syncthreads();
  const int vt = s_vt, m = s_m;
  for (int chunk = 0; chunk < L_TOK / 1024; ++chunk) {
    int l = chunk * 1024 + threadIdx.x;
    int v = vote[l];
    int iseq = (v == vt) ? 1 : 0;
    int inc_eq = block_scan_inc(iseq, sd);
    int eq_rank = s_ceq + inc_eq - iseq;           // exclusive rank among ties
    int msk = (v > vt || (iseq && eq_rank < m)) ? 1 : 0;
    int inc_m = block_scan_inc(msk, sd);
    int mp = s_cm + inc_m - msk;                   // exclusive masked count
    if (msk) ids_masked[mp] = l;
    else     ids_keep[l - mp] = l;
    __syncthreads();
    if (threadIdx.x == 1023) { s_ceq += inc_eq; s_cm += inc_m; }
    __syncthreads();
  }
}

// ---------------------------------------------------------------------------
// Kernel 4: gather kept rows into output (b128 copies).
// ---------------------------------------------------------------------------
__global__ void k_gather_keep(const float* __restrict__ x, const int* __restrict__ ids_keep,
                              float* __restrict__ out) {
  const int row = ids_keep[blockIdx.x];
  const u32x4* src = (const u32x4*)(x + (size_t)row * C_DIM);
  u32x4* dst = (u32x4*)(out + (size_t)blockIdx.x * C_DIM);
  dst[threadIdx.x] = src[threadIdx.x];            // 256 * 16B = 1024 floats
}

// ---------------------------------------------------------------------------
// Kernel 5: LayerNorm rows of global_q -> qn.
// ---------------------------------------------------------------------------
__global__ void k_ln_rows(const float* __restrict__ xin, const float* __restrict__ g,
                          const float* __restrict__ b, float* __restrict__ out) {
  const int row = blockIdx.x;
  const float* xr = xin + (size_t)row * C_DIM;
  float sum = 0.f, sq = 0.f;
  for (int c = threadIdx.x; c < C_DIM; c += 256) { float v = xr[c]; sum += v; sq += v * v; }
  #pragma unroll
  for (int o = 16; o; o >>= 1) { sum += __shfl_xor(sum, o, 32); sq += __shfl_xor(sq, o, 32); }
  __shared__ float ssum[8], ssq[8];
  __shared__ float smu, srs;
  int wid = threadIdx.x >> 5, ln = threadIdx.x & 31;
  if (ln == 0) { ssum[wid] = sum; ssq[wid] = sq; }
  __syncthreads();
  if (threadIdx.x == 0) {
    float S = 0.f, Q = 0.f;
    for (int i = 0; i < 8; ++i) { S += ssum[i]; Q += ssq[i]; }
    float mu = S * (1.f / C_DIM);
    smu = mu; srs = rsqrtf(Q * (1.f / C_DIM) - mu * mu + LN_EPS);
  }
  __syncthreads();
  for (int c = threadIdx.x; c < C_DIM; c += 256)
    out[(size_t)row * C_DIM + c] = (xr[c] - smu) * srs * g[c] + b[c];
}

// Kernel 6: q = qn @ Wq  (10x1024 @ 1024x1024, tiny).
__global__ void k_q(const float* __restrict__ qn, const float* __restrict__ Wq,
                    float* __restrict__ qv) {
  const int idx = blockIdx.x * 256 + threadIdx.x;
  const int qi = idx >> 10, e = idx & 1023;
  const float* qr = qn + (size_t)qi * C_DIM;
  float acc = 0.f;
  for (int c = 0; c < C_DIM; ++c) acc += qr[c] * Wq[(size_t)c * C_DIM + e];
  qv[idx] = acc;
}

// Kernel 7: qk[h*10+qi][c] = scale * sum_d q[qi][h*128+d] * Wk[c][h*128+d], bf16.
__global__ void k_qk(const float* __restrict__ qv, const float* __restrict__ Wkv,
                     __bf16* __restrict__ qk) {
  const int idx = blockIdx.x * 256 + threadIdx.x;
  const int r = idx >> 10, c = idx & 1023;
  const int h = r / K_Q, qi = r % K_Q;
  const float* qp = qv + (size_t)qi * C_DIM + h * D_H;
  const float* wp = Wkv + (size_t)c * (2 * C_DIM) + h * D_H;
  float acc = 0.f;
  #pragma unroll 8
  for (int d = 0; d < D_H; ++d) acc += qp[d] * wp[d];
  qk[idx] = (__bf16)(acc * QK_SCALE);
}

__global__ void k_zero_s(float* __restrict__ s_buf) {
  s_buf[blockIdx.x * 1024 + threadIdx.x] = 0.f;
}

// ---------------------------------------------------------------------------
// Kernel 8 (pass A): gather+LayerNorm 64 masked rows into LDS (bf16), emit
// transposed xn^T tile to L2-resident workspace, then WMMA dots tile:
// (80 x 1024) @ (1024 x 64) with v_wmma_f32_16x16x32_bf16.
// ---------------------------------------------------------------------------
__global__ void k_passA(const float* __restrict__ x, const int* __restrict__ ids_masked,
                        const float* __restrict__ g, const float* __restrict__ b,
                        const __bf16* __restrict__ qk, __bf16* __restrict__ xnT,
                        float* __restrict__ dots) {
  extern __shared__ char smem[];
  __bf16* lds = (__bf16*)smem;                       // 64 rows x STRH halves
  const int wave = threadIdx.x >> 5;
  const int lane = threadIdx.x & 31;
  const int lo = lane & 15, hi = lane >> 4;
  const int nb_g = blockIdx.x * 64;

  // Phase 1: gather + LN -> LDS bf16 (each wave: 16 rows; lanes split the row).
  for (int rr = 0; rr < 16; ++rr) {
    const int r_loc = wave * 16 + rr;
    const int n_g = nb_g + r_loc;
    __bf16* lrow = lds + (size_t)r_loc * STRH;
    if (n_g < KSEL) {
      const int row = ids_masked[n_g];
      const float* xr = x + (size_t)row * C_DIM;
      float vals[32];
      float sum = 0.f, sq = 0.f;
      #pragma unroll
      for (int j = 0; j < 32; ++j) { float v = xr[lane + 32 * j]; vals[j] = v; sum += v; sq += v * v; }
      #pragma unroll
      for (int o = 16; o; o >>= 1) { sum += __shfl_xor(sum, o, 32); sq += __shfl_xor(sq, o, 32); }
      const float mu = sum * (1.f / C_DIM);
      const float rs = rsqrtf(sq * (1.f / C_DIM) - mu * mu + LN_EPS);
      #pragma unroll
      for (int j = 0; j < 32; ++j) {
        int c = lane + 32 * j;
        lrow[c] = (__bf16)((vals[j] - mu) * rs * g[c] + b[c]);
      }
    } else {
      #pragma unroll
      for (int j = 0; j < 32; ++j) lrow[lane + 32 * j] = (__bf16)0.f;
    }
  }
  __syncthreads();

  // Phase 2: coalesced write of transposed tile: xnT[feature][token].
  for (int c = wave; c < C_DIM; c += 4) {
    const int n0 = lane * 2;
    union { __bf16 h[2]; unsigned u; } pk;
    pk.h[0] = lds[(size_t)(n0    ) * STRH + c];
    pk.h[1] = lds[(size_t)(n0 + 1) * STRH + c];
    *(unsigned*)((char*)xnT + ((size_t)c * NPAD + nb_g + n0) * 2) = pk.u;
  }

  // Phase 3: WMMA dots tile. Each wave owns a 16-col subtile; 5 M-tiles of 16.
  v8f acc[5] = {};
  const int ncol_loc = wave * 16;
  for (int k0 = 0; k0 < C_DIM; k0 += 32) {
    FragU bf;
    const __bf16* bb = lds + (size_t)(ncol_loc + lo) * STRH + k0 + hi * 8;
    bf.u.a = *(const u32x4*)bb;                      // ds_load_b128
    bf.u.b = *(const u32x4*)(bb + 16);
    #pragma unroll
    for (int mt = 0; mt < 5; ++mt) {
      FragU af;
      const __bf16* ab = qk + (size_t)(mt * 16 + lo) * C_DIM + k0 + hi * 8;
      af.u.a = *(const u32x4*)ab;                    // global_load_b128 (L2-hot)
      af.u.b = *(const u32x4*)(ab + 16);
      acc[mt] = __builtin_amdgcn_wmma_f32_16x16x32_bf16(
          false, af.f, false, bf.f, (short)0, acc[mt], false, false);
    }
  }
  const int ncol_g = nb_g + ncol_loc + lo;
  #pragma unroll
  for (int mt = 0; mt < 5; ++mt)
    #pragma unroll
    for (int r = 0; r < 8; ++r)
      dots[(size_t)(mt * 16 + hi * 8 + r) * NPAD + ncol_g] = acc[mt][r];
}

// ---------------------------------------------------------------------------
// Kernel 9: row softmax over KSEL, emit bf16 weights (padded tail zeroed so
// pass C can run the full padded K range unconditionally).
// ---------------------------------------------------------------------------
__global__ void k_softmax(const float* __restrict__ dots, __bf16* __restrict__ wbf) {
  const int row = blockIdx.x;
  const float* dr = dots + (size_t)row * NPAD;
  __shared__ float red[8];
  __shared__ float bcast;
  const int wid = threadIdx.x >> 5, ln = threadIdx.x & 31;

  float mx = -3.4e38f;
  for (int n = threadIdx.x; n < KSEL; n += 256) mx = fmaxf(mx, dr[n]);
  #pragma unroll
  for (int o = 16; o; o >>= 1) mx = fmaxf(mx, __shfl_xor(mx, o, 32));
  if (ln == 0) red[wid] = mx;
  __syncthreads();
  if (threadIdx.x == 0) { float m = red[0]; for (int i = 1; i < 8; ++i) m = fmaxf(m, red[i]); bcast = m; }
  __syncthreads();
  const float M = bcast;
  __syncthreads();

  float sum = 0.f;
  for (int n = threadIdx.x; n < KSEL; n += 256) sum += __expf(dr[n] - M);
  #pragma unroll
  for (int o = 16; o; o >>= 1) sum += __shfl_xor(sum, o, 32);
  if (ln == 0) red[wid] = sum;
  __syncthreads();
  if (threadIdx.x == 0) { float t = 0.f; for (int i = 0; i < 8; ++i) t += red[i]; bcast = t; }
  __syncthreads();
  const float inv = 1.f / bcast;

  __bf16* wr = wbf + (size_t)row * NPAD;
  for (int n = threadIdx.x; n < KSEL; n += 256) wr[n] = (__bf16)(__expf(dr[n] - M) * inv);
  for (int n = KSEL + threadIdx.x; n < NPAD; n += 256) wr[n] = (__bf16)0.f;
}

// ---------------------------------------------------------------------------
// Kernel 10 (pass C): s = w @ xn  (80 x NPAD) @ (NPAD x 1024), WMMA bf16,
// K split across blockIdx.y with float atomics into s.
// ---------------------------------------------------------------------------
__global__ void k_s(const __bf16* __restrict__ wbf, const __bf16* __restrict__ xnT,
                    float* __restrict__ s_buf) {
  const int wave = threadIdx.x >> 5;
  const int lane = threadIdx.x & 31;
  const int lo = lane & 15, hi = lane >> 4;
  const int nf = blockIdx.x * 64 + wave * 16;        // output feature base
  const int kbase = blockIdx.y * (NPAD / 4);
  v8f acc[5] = {};
  for (int k0 = kbase; k0 < kbase + NPAD / 4; k0 += 32) {
    FragU bf;
    const __bf16* bb = xnT + (size_t)(nf + lo) * NPAD + k0 + hi * 8;
    bf.u.a = *(const u32x4*)bb;
    bf.u.b = *(const u32x4*)(bb + 16);
    #pragma unroll
    for (int mt = 0; mt < 5; ++mt) {
      FragU af;
      const __bf16* ab = wbf + (size_t)(mt * 16 + lo) * NPAD + k0 + hi * 8;
      af.u.a = *(const u32x4*)ab;
      af.u.b = *(const u32x4*)(ab + 16);
      acc[mt] = __builtin_amdgcn_wmma_f32_16x16x32_bf16(
          false, af.f, false, bf.f, (short)0, acc[mt], false, false);
    }
  }
  #pragma unroll
  for (int mt = 0; mt < 5; ++mt)
    #pragma unroll
    for (int r = 0; r < 8; ++r)
      atomicAdd(&s_buf[(size_t)(mt * 16 + hi * 8 + r) * C_DIM + nf + lo], acc[mt][r]);
}

// Kernel 11: out[qi][h*128+d] = sum_c s[h*10+qi][c] * Wv[c][h*128+d].
__global__ void k_outa(const float* __restrict__ s_buf, const float* __restrict__ Wkv,
                       float* __restrict__ outa) {
  const int idx = blockIdx.x * 256 + threadIdx.x;
  const int qi = idx >> 10, e = idx & 1023;
  const int h = e >> 7;
  const float* sr = s_buf + (size_t)(h * K_Q + qi) * C_DIM;
  float acc = 0.f;
  for (int c = 0; c < C_DIM; ++c) acc += sr[c] * Wkv[(size_t)c * (2 * C_DIM) + C_DIM + e];
  outa[idx] = acc;
}

// Kernel 12: z = out @ Wo + bo -> final rows of d_out.
__global__ void k_z(const float* __restrict__ outa, const float* __restrict__ Wo,
                    const float* __restrict__ bo, float* __restrict__ out) {
  const int idx = blockIdx.x * 256 + threadIdx.x;
  const int qi = idx >> 10, c = idx & 1023;
  const float* orow = outa + (size_t)qi * C_DIM;
  float acc = bo[c];
  for (int e = 0; e < C_DIM; ++e) acc += orow[e] * Wo[(size_t)e * C_DIM + c];
  out[(size_t)(LKEEP + qi) * C_DIM + c] = acc;
}

// ---------------------------------------------------------------------------
extern "C" void kernel_launch(void* const* d_in, const int* in_sizes, int n_in,
                              void* d_out, int out_size, void* d_ws, size_t ws_size,
                              hipStream_t stream) {
  (void)in_sizes; (void)n_in; (void)out_size; (void)ws_size;
  const float* x    = (const float*)d_in[0];
  const float* attn = (const float*)d_in[1];
  const float* ln_g = (const float*)d_in[2];
  const float* ln_b = (const float*)d_in[3];
  const float* Wq   = (const float*)d_in[4];
  const float* Wkv  = (const float*)d_in[5];
  const float* Wo   = (const float*)d_in[6];
  const float* bo   = (const float*)d_in[7];
  const float* gq   = (const float*)d_in[8];
  float* out = (float*)d_out;

  char* w = (char*)d_ws;
  auto take = [&](size_t bytes) -> char* {
    char* p = w; w += (bytes + 255) & ~(size_t)255; return p;
  };
  unsigned* thresh   = (unsigned*)take((size_t)N_H * 4);
  int*      c_hist   = (int*)     take(16 * 4);
  int*      vote     = (int*)     take((size_t)L_TOK * 4);
  int*      ids_keep = (int*)     take((size_t)LKEEP * 4);
  int*      ids_mask = (int*)     take((size_t)NPAD * 4);
  float*    qn       = (float*)   take((size_t)K_Q * C_DIM * 4);
  float*    qv       = (float*)   take((size_t)K_Q * C_DIM * 4);
  __bf16*   qk       = (__bf16*)  take((size_t)80 * C_DIM * 2);
  float*    s_buf    = (float*)   take((size_t)80 * C_DIM * 4);
  float*    outa     = (float*)   take((size_t)K_Q * C_DIM * 4);
  float*    dots     = (float*)   take((size_t)80 * NPAD * 4);
  __bf16*   wbf      = (__bf16*)  take((size_t)80 * NPAD * 2);
  __bf16*   xnT      = (__bf16*)  take((size_t)C_DIM * NPAD * 2);   // ~54 MB, L2-resident

  k_radix      <<<N_H, 256, 0, stream>>>(attn, thresh, c_hist);
  k_vote       <<<L_TOK / 256, 256, 0, stream>>>(attn, thresh, vote, c_hist);
  k_compact    <<<1, 1024, 0, stream>>>(vote, c_hist, ids_keep, ids_mask);
  k_gather_keep<<<LKEEP, 256, 0, stream>>>(x, ids_keep, out);
  k_ln_rows    <<<K_Q, 256, 0, stream>>>(gq, ln_g, ln_b, qn);
  k_q          <<<(K_Q * C_DIM) / 256, 256, 0, stream>>>(qn, Wq, qv);
  k_qk         <<<(80 * C_DIM) / 256, 256, 0, stream>>>(qv, Wkv, qk);
  k_zero_s     <<<80, 1024, 0, stream>>>(s_buf);
  k_passA      <<<NPAD / 64, 128, 64 * 2064, stream>>>(x, ids_mask, ln_g, ln_b, qk, xnT, dots);
  k_softmax    <<<80, 256, 0, stream>>>(dots, wbf);
  k_s          <<<dim3(16, 4), 128, 0, stream>>>(wbf, xnT, s_buf);
  k_outa       <<<(K_Q * C_DIM) / 256, 256, 0, stream>>>(s_buf, Wkv, outa);
  k_z          <<<(K_Q * C_DIM) / 256, 256, 0, stream>>>(outa, Wo, bo, out);
}
